// DMGCN_29609504538897
// MI455X (gfx1250) — matrix operand order repaired and checked by reading, not compiled, not run on hardware
//
#include <hip/hip_runtime.h>

typedef __attribute__((ext_vector_type(16))) __bf16 v16bf;
typedef __attribute__((ext_vector_type(8)))  float  v8f;

#define NNODES 100000
#define NEDGES 400000
#define NGRAPH 2000
#define DIM    128
#define DE     428      // 128 + 300 RBF
#define DEP    448      // DE padded to multiple of 32 (14 K-steps)
#define SE     456      // LDS row stride for DEP tiles (228 DW: conflict-free b128 rows)
#define SH     136      // LDS row stride for DIM tiles (68 DW: conflict-free b128 rows)
#define NCONV  3
#define WAVES  4        // waves per block; lockstep waves share weight stream in WGP$

// per-wave LDS slice (elements): eT + ee1T (DEP tiles) + hsT + hn1T (DIM tiles)
#define EL_E   (16 * SE)
#define EL_H   (16 * SH)
#define WSLICE (2 * EL_E + 2 * EL_H)

// ---------------- WMMA fragment loaders (CDNA5 16-bit layouts) ----------------
// A (16x32 bf16): lane m = lane&15, khalf = lane>>4; elements 0..7 -> K = kt*32+8*khalf+e,
// elements 8..15 -> K = kt*32+16+8*khalf+(e-8): two 16B chunks 32B apart.
__device__ __forceinline__ v16bf load_frag_a(const __bf16* p) {
  union { uint4 q[2]; v16bf v; } u;
  u.q[0] = *reinterpret_cast<const uint4*>(p);
  u.q[1] = *reinterpret_cast<const uint4*>(p + 16);
  return u.v;
}
// B (32x16 bf16): lane n = lane&15, khalf = lane>>4; elements 0..7 -> K = kt*32+8*khalf+e,
// elements 8..15 -> K = kt*32+16+8*khalf+(e-8): two 16B chunks 32B apart (matches A).
__device__ __forceinline__ v16bf load_frag_b(const __bf16* p) {
  union { uint4 q[2]; v16bf v; } u;
  u.q[0] = *reinterpret_cast<const uint4*>(p);
  u.q[1] = *reinterpret_cast<const uint4*>(p + 16);
  return u.v;
}

// C = A(16xK from LDS) x W^T(bf16 row-major [n][kpad] in global/L2), f32 accum.
// Epilogue sees the documented C layout: lane col = lane&15, rows m = (lane>>4)*8 + r.
template <typename Epi>
__device__ __forceinline__ void wmma_gemm(const __bf16* aT, int astride,
                                          const __bf16* W, int kpad,
                                          int ntiles, int ktiles, int lane, Epi epi) {
  const int col  = lane & 15;
  const int half = lane >> 4;
  const int rowoff = half * 8;
  for (int nt = 0; nt < ntiles; ++nt) {
    v8f acc = {0.f, 0.f, 0.f, 0.f, 0.f, 0.f, 0.f, 0.f};
    for (int kt = 0; kt < ktiles; ++kt) {
      v16bf a = load_frag_a(aT + col * astride + kt * 32 + half * 8);
      v16bf b = load_frag_b(W + (long)(nt * 16 + col) * kpad + kt * 32 + half * 8);
      acc = __builtin_amdgcn_wmma_f32_16x16x32_bf16(false, a, false, b,
                                                    (short)0, acc, false, false);
    }
    epi(nt, col, rowoff, acc);
  }
}

// -------- edge-conv kernel: 4 lockstep waves/block, one 16-edge tile per wave --------
__global__ __launch_bounds__(32 * WAVES) void edge_conv_kernel(
    const float* __restrict__ h_in, float* __restrict__ h_out,
    const int* __restrict__ srcI, const int* __restrict__ dstI,
    const int* __restrict__ etype, const float* __restrict__ dist,
    const float* __restrict__ edge_emb,
    const __bf16* __restrict__ wn1, const __bf16* __restrict__ wn2,
    const __bf16* __restrict__ wc,  const __bf16* __restrict__ we1,
    const __bf16* __restrict__ we2,
    const float* __restrict__ bn1, const float* __restrict__ bn2,
    const float* __restrict__ be1, const float* __restrict__ be2,
    const float* __restrict__ bc)
{
  __shared__ __align__(16) __bf16 smem[WAVES * WSLICE];   // 151,552 B
  const int lane = threadIdx.x & 31;
  const int wave = threadIdx.x >> 5;
  __bf16* eT   = smem + wave * WSLICE;       // edge features (bf16, zero-padded)
  __bf16* ee1T = eT + EL_E;                  // relu(e @ We1^T)
  __bf16* hsT  = ee1T + EL_E;                // h[src]; later reused for hn
  __bf16* hn1T = hsT + EL_H;                 // relu(hs@Wn1^T); later reused for prod
  const long eb = ((long)blockIdx.x * WAVES + wave) * 16;

  // stage gathered source-node features
  for (int r = 0; r < 16; ++r) {
    const float* hrow = h_in + (long)srcI[eb + r] * DIM;
    for (int c = lane; c < DIM; c += 32)
      hsT[r * SH + c] = (__bf16)hrow[c];
  }
  // stage edge features: [edge_emb | rbf(dist) | zero-pad]
  const float inv_gap = 299.0f / 30.0f;   // 1 / (30/299)
  for (int r = 0; r < 16; ++r) {
    const float* erow = edge_emb + (long)etype[eb + r] * DIM;
    const float d = dist[eb + r];
    for (int c = lane; c < DEP; c += 32) {
      float v = 0.f;
      if (c < DIM) {
        v = erow[c];
      } else if (c < DE) {
        float t = d - (30.0f / 299.0f) * (float)(c - DIM);
        v = __expf(-t * t * inv_gap);
      }
      eT[r * SE + c] = (__bf16)v;
    }
  }
  __syncthreads();

  // hn1 = relu(hs @ Wn1^T + bn1)
  wmma_gemm(hsT, SH, wn1, DIM, 8, 4, lane,
    [&](int nt, int col, int rowoff, v8f acc) {
      float b = bn1[nt * 16 + col];
      #pragma unroll
      for (int r = 0; r < 8; ++r)
        hn1T[(rowoff + r) * SH + nt * 16 + col] = (__bf16)fmaxf(acc[r] + b, 0.f);
    });
  __syncthreads();

  // hn = hn1 @ Wn2^T + bn2   (written into hsT, which is now free)
  wmma_gemm(hn1T, SH, wn2, DIM, 8, 4, lane,
    [&](int nt, int col, int rowoff, v8f acc) {
      float b = bn2[nt * 16 + col];
      #pragma unroll
      for (int r = 0; r < 8; ++r)
        hsT[(rowoff + r) * SH + nt * 16 + col] = (__bf16)(acc[r] + b);
    });
  __syncthreads();

  // ee1 = relu(e @ We1^T + be1)  (padded cols: zero weights + zero bias -> stay 0)
  wmma_gemm(eT, SE, we1, DEP, 28, 14, lane,
    [&](int nt, int col, int rowoff, v8f acc) {
      int n = nt * 16 + col;
      float b = (n < DE) ? be1[n] : 0.f;
      #pragma unroll
      for (int r = 0; r < 8; ++r)
        ee1T[(rowoff + r) * SE + n] = (__bf16)fmaxf(acc[r] + b, 0.f);
    });
  __syncthreads();

  // prod = (ee1 @ We2^T + be2) * hn     (written into hn1T, now free)
  wmma_gemm(ee1T, SE, we2, DEP, 8, 14, lane,
    [&](int nt, int col, int rowoff, v8f acc) {
      int n = nt * 16 + col;
      float b = be2[n];
      #pragma unroll
      for (int r = 0; r < 8; ++r) {
        int m = rowoff + r;
        float hnv = (float)hsT[m * SH + n];
        hn1T[m * SH + n] = (__bf16)((acc[r] + b) * hnv);
      }
    });
  __syncthreads();

  // m = tanh(prod @ Wc^T + bc); scatter-add into h_out[dst]
  wmma_gemm(hn1T, SH, wc, DIM, 8, 4, lane,
    [&](int nt, int col, int rowoff, v8f acc) {
      int n = nt * 16 + col;
      float b = bc[n];
      #pragma unroll
      for (int r = 0; r < 8; ++r) {
        int m = rowoff + r;
        float v = tanhf(acc[r] + b);
        unsafeAtomicAdd(&h_out[(long)dstI[eb + m] * DIM + n], v);
      }
    });
}

// ---------------- small helper kernels ----------------
__global__ void h_init_kernel(const int* __restrict__ Z, const float* __restrict__ node_emb,
                              float* __restrict__ h, long total) {
  long i = (long)blockIdx.x * blockDim.x + threadIdx.x;
  if (i >= total) return;
  long n = i >> 7; int c = (int)(i & 127);
  h[i] = node_emb[(long)Z[n] * DIM + c];
}

__global__ void copy_kernel(const float* __restrict__ a, float* __restrict__ b, long total) {
  long i = (long)blockIdx.x * blockDim.x + threadIdx.x;
  if (i < total) b[i] = a[i];
}

__global__ void zero_kernel(float* __restrict__ p, long total) {
  long i = (long)blockIdx.x * blockDim.x + threadIdx.x;
  if (i < total) p[i] = 0.f;
}

// f32 [rows x cols] row-major -> bf16 [rpad x cpad] row-major, zero-padded
__global__ void convert_pad_kernel(const float* __restrict__ src, __bf16* __restrict__ dst,
                                   int rows, int cols, int rpad, int cpad) {
  long i = (long)blockIdx.x * blockDim.x + threadIdx.x;
  long total = (long)rpad * cpad;
  if (i >= total) return;
  int r = (int)(i / cpad);
  int c = (int)(i - (long)r * cpad);
  float v = (r < rows && c < cols) ? src[(long)r * cols + c] : 0.f;
  dst[i] = (__bf16)v;
}

// readout: r = relu(h@Wr1^T+br1)@Wr2^T+br2 ; out[g] += r    (one block per node)
__global__ __launch_bounds__(128) void readout_kernel(
    const float* __restrict__ h, const float* __restrict__ Wr1,
    const float* __restrict__ br1, const float* __restrict__ Wr2,
    const float* __restrict__ br2, const int* __restrict__ gid,
    float* __restrict__ out) {
  __shared__ float row[DIM];
  __shared__ float red[DIM];
  int n = blockIdx.x, t = threadIdx.x;
  row[t] = h[(long)n * DIM + t];
  __syncthreads();
  float s = 0.f;
  #pragma unroll 8
  for (int k = 0; k < DIM; ++k) s += row[k] * Wr1[t * DIM + k];
  s = fmaxf(s + br1[t], 0.f) * Wr2[t];
  red[t] = s;
  __syncthreads();
  for (int off = 64; off > 0; off >>= 1) {
    if (t < off) red[t] += red[t + off];
    __syncthreads();
  }
  if (t == 0) unsafeAtomicAdd(&out[gid[n]], red[0] + br2[0]);
}

// ---------------- host side ----------------
extern "C" void kernel_launch(void* const* d_in, const int* in_sizes, int n_in,
                              void* d_out, int out_size, void* d_ws, size_t ws_size,
                              hipStream_t stream) {
  (void)in_sizes; (void)n_in; (void)out_size; (void)ws_size;
  const int*   Z        = (const int*)  d_in[0];
  const int*   etype    = (const int*)  d_in[1];
  const float* dist     = (const float*)d_in[2];
  const int*   srcI     = (const int*)  d_in[3];
  const int*   dstI     = (const int*)  d_in[4];
  const int*   gid      = (const int*)  d_in[5];
  const float* node_emb = (const float*)d_in[6];
  const float* edge_emb = (const float*)d_in[7];
  const float* Wn1 = (const float*)d_in[8];
  const float* bn1 = (const float*)d_in[9];
  const float* Wn2 = (const float*)d_in[10];
  const float* bn2 = (const float*)d_in[11];
  const float* We1 = (const float*)d_in[12];
  const float* be1 = (const float*)d_in[13];
  const float* We2 = (const float*)d_in[14];
  const float* be2 = (const float*)d_in[15];
  const float* Wc  = (const float*)d_in[16];
  const float* bc  = (const float*)d_in[17];
  const float* Wr1 = (const float*)d_in[18];
  const float* br1 = (const float*)d_in[19];
  const float* Wr2 = (const float*)d_in[20];
  const float* br2 = (const float*)d_in[21];
  float* out = (float*)d_out;

  // workspace layout (~104 MB): two f32 node buffers + bf16 padded weights
  char* ws = (char*)d_ws;
  size_t off = 0;
  auto alloc = [&](size_t bytes) -> void* {
    off = (off + 255) & ~size_t(255);
    void* p = ws + off;
    off += bytes;
    return p;
  };
  float*  hA    = (float*) alloc((size_t)NNODES * DIM * 4);
  float*  hB    = (float*) alloc((size_t)NNODES * DIM * 4);
  __bf16* wn1bf = (__bf16*)alloc((size_t)NCONV * DIM * DIM * 2);
  __bf16* wn2bf = (__bf16*)alloc((size_t)NCONV * DIM * DIM * 2);
  __bf16* wcbf  = (__bf16*)alloc((size_t)NCONV * DIM * DIM * 2);
  __bf16* we1bf = (__bf16*)alloc((size_t)NCONV * DEP * DEP * 2);
  __bf16* we2bf = (__bf16*)alloc((size_t)NCONV * DIM * DEP * 2);

  auto conv = [&](const float* s, __bf16* d, int rows, int cols, int rpad, int cpad) {
    long tot = (long)rpad * cpad;
    convert_pad_kernel<<<(tot + 255) / 256, 256, 0, stream>>>(s, d, rows, cols, rpad, cpad);
  };
  for (int i = 0; i < NCONV; ++i) {
    conv(Wn1 + (size_t)i * DIM * DIM, wn1bf + (size_t)i * DIM * DIM, DIM, DIM, DIM, DIM);
    conv(Wn2 + (size_t)i * DIM * DIM, wn2bf + (size_t)i * DIM * DIM, DIM, DIM, DIM, DIM);
    conv(Wc  + (size_t)i * DIM * DIM, wcbf  + (size_t)i * DIM * DIM, DIM, DIM, DIM, DIM);
    conv(We1 + (size_t)i * DE  * DE,  we1bf + (size_t)i * DEP * DEP, DE,  DE,  DEP, DEP);
    conv(We2 + (size_t)i * DIM * DE,  we2bf + (size_t)i * DIM * DEP, DIM, DE,  DIM, DEP);
  }

  const long hTot = (long)NNODES * DIM;
  h_init_kernel<<<(hTot + 255) / 256, 256, 0, stream>>>(Z, node_emb, hA, hTot);

  float* hin = hA;
  float* hout = hB;
  for (int i = 0; i < NCONV; ++i) {
    copy_kernel<<<(hTot + 255) / 256, 256, 0, stream>>>(hin, hout, hTot);
    edge_conv_kernel<<<NEDGES / (16 * WAVES), 32 * WAVES, 0, stream>>>(
        hin, hout, srcI, dstI, etype, dist, edge_emb,
        wn1bf + (size_t)i * DIM * DIM, wn2bf + (size_t)i * DIM * DIM,
        wcbf  + (size_t)i * DIM * DIM, we1bf + (size_t)i * DEP * DEP,
        we2bf + (size_t)i * DIM * DEP,
        bn1 + i * DIM, bn2 + i * DIM, be1 + i * DE, be2 + i * DIM, bc + i * DIM);
    float* t = hin; hin = hout; hout = t;
  }

  zero_kernel<<<(NGRAPH + 255) / 256, 256, 0, stream>>>(out, NGRAPH);
  readout_kernel<<<NNODES, 128, 0, stream>>>(hin, Wr1, br1, Wr2, br2, gid, out);
}
